// mowLSTM_31920196944208
// MI455X (gfx1250) — matrix operand order repaired and checked by reading, not compiled, not run on hardware
//
#include <hip/hip_runtime.h>
#include <hip/hip_bf16.h>

// ---------------------------------------------------------------------------
// MoO-LSTM for MI455X (gfx1250), bf16 WMMA (16x16x32, f32 accum).
//
// Key design (roofline): per-sample mixed hidden weights Whm [B,4H,H] would
// force ~1 GB of HBM traffic per timestep (no batch reuse) -> ~11.5 ms at
// 23.3 TB/s. Instead we keep the 4 expert weight sets in bf16 (33.5 MB,
// L2-resident in the 192 MB L2) and mix the *projection outputs* with coef
// in the epilogue: 4x FLOPs but WMMA-bound (~1-2 us/step) instead of
// HBM-bound. Recurrence runs in one persistent kernel with an atomic grid
// barrier to avoid 256 sequential kernel-launch latencies.
//
// Fragment layouts (per CDNA5 ISA, wave32):
//   A (16x32 bf16): lane l -> row l&15; two 16B chunks at K = (l>=16)*8 and
//                   16+(l>=16)*8.
//   B (32x16 bf16): lane l -> col l&15; two 16B chunks of [N,K] storage
//                   at K = (l>=16)*8 and 16+(l>=16)*8  ==  weights' natural [out,in] layout.
//   C/D (16x16 f32): VGPR v, lane l -> row v + 8*(l>=16), col l&15.
// ---------------------------------------------------------------------------

#define T_ 256
#define B_ 64
#define I_ 1024
#define H_ 1024
#define G_ 4096   // 4*H
#define E_ 4
#define C_ 128

typedef __attribute__((ext_vector_type(16))) __bf16 v16bf;
typedef __attribute__((ext_vector_type(8)))  float  v8f;

union ABFrag { v16bf v; uint4 q[2]; };

__device__ __forceinline__ uint4 ldg128(const __bf16* p) {
  return *reinterpret_cast<const uint4*>(p);
}

__device__ __forceinline__ v8f wmma_bf16(v16bf a, v16bf b, v8f c) {
  // (neg_a, A, neg_b, B, c_mod, C, reuse_a, reuse_b)
  return __builtin_amdgcn_wmma_f32_16x16x32_bf16(false, a, false, b, (short)0, c,
                                                 false, false);
}

__device__ __forceinline__ v8f zero8() {
  v8f z;
#pragma unroll
  for (int i = 0; i < 8; ++i) z[i] = 0.0f;
  return z;
}

__device__ __forceinline__ float sigm(float x) {
  return 1.0f / (1.0f + __expf(-x));
}

// --------------------------- fp32 -> bf16 convert ---------------------------
__global__ void k_f32_to_bf16(const float* __restrict__ src,
                              __bf16* __restrict__ dst, int n) {
  int i = blockIdx.x * blockDim.x + threadIdx.x;
  int stride = gridDim.x * blockDim.x;
  for (; i < n; i += stride) dst[i] = (__bf16)src[i];
}

// ------------------- mixed biases, state init, barrier init -----------------
__global__ void k_init(const float* __restrict__ coef,
                       const float* __restrict__ bi, const float* __restrict__ bh,
                       const float* __restrict__ bo,
                       const float* __restrict__ h0, const float* __restrict__ c0,
                       float* __restrict__ biasg,   // [B][G] : coef@(bi+bh)
                       float* __restrict__ boc,     // [B][C] : coef@bo
                       __bf16* __restrict__ hbf,    // [B][H]
                       float* __restrict__ cbuf,    // [B][H]
                       int* __restrict__ bar) {
  int idx = blockIdx.x * blockDim.x + threadIdx.x;  // B_*G_ threads
  if (idx < B_ * G_) {
    int b = idx / G_, o = idx % G_;
    float s = 0.f;
#pragma unroll
    for (int e = 0; e < E_; ++e) s += coef[b * E_ + e] * (bi[e * G_ + o] + bh[e * G_ + o]);
    biasg[idx] = s;
  }
  if (idx < B_ * C_) {
    int b = idx / C_, c = idx % C_;
    float s = 0.f;
#pragma unroll
    for (int e = 0; e < E_; ++e) s += coef[b * E_ + e] * bo[e * C_ + c];
    boc[idx] = s;
  }
  if (idx < B_ * H_) {
    hbf[idx]  = (__bf16)h0[idx];
    cbuf[idx] = c0[idx];
  }
  if (idx < 2) bar[idx] = 0;
}

// -------- input projection: xi[m][o] = biasg + sum_e coef*(x@Wi_e^T) --------
// M = T*B (m = t*B+b), N = G, K = I. One 16x16 tile per wave, 4 expert accs.
__global__ void __launch_bounds__(256) k_gemm_xi(
    const __bf16* __restrict__ xbf,   // [T*B][I] row-major
    const __bf16* __restrict__ wibf,  // [E][G][I]  ([N,K] per expert)
    const float*  __restrict__ coef,  // [B][E]
    const float*  __restrict__ biasg, // [B][G]
    float*        __restrict__ xi) {  // [T*B][G]
  const int lane = threadIdx.x & 31;
  const int hi   = lane >> 4;        // K-half selector
  const int ln   = lane & 15;
  const int wave = (blockIdx.x * blockDim.x + threadIdx.x) >> 5;
  const int NT = G_ / 16;            // 256
  const int MT = (T_ * B_) / 16;     // 1024
  if (wave >= MT * NT) return;
  const int m0 = (wave / NT) * 16;
  const int n0 = (wave % NT) * 16;
  const int colB = n0 + ln;

  const __bf16* arow = xbf + (size_t)(m0 + ln) * I_;
  const __bf16* br[E_];
#pragma unroll
  for (int e = 0; e < E_; ++e) br[e] = wibf + ((size_t)e * G_ + colB) * I_;

  v8f acc[E_];
#pragma unroll
  for (int e = 0; e < E_; ++e) acc[e] = zero8();

  for (int k0 = 0; k0 < I_; k0 += 32) {
    ABFrag a;
    a.q[0] = ldg128(arow + k0 + hi * 8);
    a.q[1] = ldg128(arow + k0 + 16 + hi * 8);
    // locality 3 -> WGP-scope prefetch (pull into all cache levels)
    if (k0 + 32 < I_) __builtin_prefetch(arow + k0 + 32, 0, 3);
#pragma unroll
    for (int e = 0; e < E_; ++e) {
      const __bf16* p = br[e] + k0 + hi * 8;
      ABFrag b;
      b.q[0] = ldg128(p);
      b.q[1] = ldg128(p + 16);
      acc[e] = wmma_bf16(a.v, b.v, acc[e]);
    }
  }
#pragma unroll
  for (int v = 0; v < 8; ++v) {
    int m = m0 + v + hi * 8;
    int b = m & (B_ - 1);            // m = t*B + b, B power of two
    float s = biasg[b * G_ + colB];
#pragma unroll
    for (int e = 0; e < E_; ++e) s += coef[b * E_ + e] * acc[e][v];
    xi[(size_t)m * G_ + colB] = s;
  }
}

// ------------------------------ grid barrier -------------------------------
__device__ __forceinline__ void grid_sync(int* cnt, int* gen, int nblk, int target) {
  __syncthreads();
  if (threadIdx.x == 0) {
    __threadfence();
    if (atomicAdd(cnt, 1) == nblk - 1) {
      atomicExch(cnt, 0);            // reset BEFORE releasing waiters
      __threadfence();
      atomicExch(gen, target);
    } else {
      while (atomicAdd(gen, 0) < target) __builtin_amdgcn_s_sleep(1);
      __threadfence();
    }
  }
  __syncthreads();
}

#define LSTM_BLOCKS 64

// ------------------------- persistent recurrent kernel ----------------------
__global__ void __launch_bounds__(256) k_lstm(
    const __bf16* __restrict__ whbf,  // [E][G][H] (L2-resident, 33.5 MB)
    const float*  __restrict__ coef,  // [B][E]
    const float*  __restrict__ xi,    // [T*B][G] (biases folded in)
    __bf16*       __restrict__ hbf,   // [B][H] current h (bf16)
    float*        __restrict__ cbuf,  // [B][H] cell state (fp32)
    float*        __restrict__ gates, // [B][G] scratch
    __bf16*       __restrict__ hsbf,  // [T*B][H] all h's (bf16) for h2o GEMM
    float*        __restrict__ outHT, // [B*H]
    float*        __restrict__ outCT, // [B*H]
    int*          __restrict__ bar) {
  const int lane = threadIdx.x & 31;
  const int hi   = lane >> 4;
  const int ln   = lane & 15;
  const int gw   = blockIdx.x * 8 + (threadIdx.x >> 5);
  const int nwaves   = LSTM_BLOCKS * 8;              // 512
  const int tid      = blockIdx.x * blockDim.x + threadIdx.x;
  const int nthreads = LSTM_BLOCKS * 256;
  const int NTILES = (B_ / 16) * (G_ / 16);          // 1024

  int bc = 0;
  for (int t = 0; t < T_; ++t) {
    // ---- GEMM phase: gates[b][o] = xi[t] + sum_e coef[b,e]*(h @ Wh_e^T) ----
    for (int tile = gw; tile < NTILES; tile += nwaves) {
      const int m0 = (tile >> 8) * 16;               // G_/16 == 256 N-tiles
      const int n0 = (tile & 255) * 16;
      const int colB = n0 + ln;

      const __bf16* arow = hbf + (size_t)(m0 + ln) * H_;
      const __bf16* br[E_];
#pragma unroll
      for (int e = 0; e < E_; ++e) br[e] = whbf + ((size_t)e * G_ + colB) * H_;

      v8f acc[E_];
#pragma unroll
      for (int e = 0; e < E_; ++e) acc[e] = zero8();

      for (int k0 = 0; k0 < H_; k0 += 32) {
        ABFrag a;
        a.q[0] = ldg128(arow + k0 + hi * 8);
        a.q[1] = ldg128(arow + k0 + 16 + hi * 8);
        if (k0 + 32 < H_) __builtin_prefetch(arow + k0 + 32, 0, 3);
#pragma unroll
        for (int e = 0; e < E_; ++e) {
          const __bf16* p = br[e] + k0 + hi * 8;
          ABFrag b;
          b.q[0] = ldg128(p);
          b.q[1] = ldg128(p + 16);
          acc[e] = wmma_bf16(a.v, b.v, acc[e]);
        }
      }
#pragma unroll
      for (int v = 0; v < 8; ++v) {
        int b = m0 + v + hi * 8;                     // row == batch index
        float s = xi[((size_t)t * B_ + b) * G_ + colB];
#pragma unroll
        for (int e = 0; e < E_; ++e) s += coef[b * E_ + e] * acc[e][v];
        gates[(size_t)b * G_ + colB] = s;
      }
    }
    ++bc;
    grid_sync(bar, bar + 1, LSTM_BLOCKS, bc);

    // ---- elementwise LSTM update ----
    for (int idx = tid; idx < B_ * H_; idx += nthreads) {
      int b  = idx >> 10;                            // H_ == 1024
      int hh = idx & (H_ - 1);
      const float* gb = gates + (size_t)b * G_;
      float ig = gb[hh];
      float fg = gb[H_ + hh];
      float gg = gb[2 * H_ + hh];
      float og = gb[3 * H_ + hh];
      float cn = sigm(fg) * cbuf[idx] + sigm(ig) * tanhf(gg);
      float hn = sigm(og) * tanhf(cn);
      cbuf[idx] = cn;
      hbf[idx]  = (__bf16)hn;
      hsbf[((size_t)t * B_ + b) * H_ + hh] = (__bf16)hn;
      if (t == T_ - 1) { outHT[idx] = hn; outCT[idx] = cn; }
    }
    ++bc;
    grid_sync(bar, bar + 1, LSTM_BLOCKS, bc);
  }
}

// -------- output projection: out[m][c] = boc + sum_e coef*(h@Wo_e^T) --------
__global__ void __launch_bounds__(256) k_gemm_out(
    const __bf16* __restrict__ hsbf,  // [T*B][H]
    const __bf16* __restrict__ wobf,  // [E][C][H]
    const float*  __restrict__ coef,
    const float*  __restrict__ boc,   // [B][C]
    float*        __restrict__ out) { // [T*B][C] == [T,B,C]
  const int lane = threadIdx.x & 31;
  const int hi   = lane >> 4;
  const int ln   = lane & 15;
  const int wave = (blockIdx.x * blockDim.x + threadIdx.x) >> 5;
  const int NT = C_ / 16;            // 8
  const int MT = (T_ * B_) / 16;     // 1024
  if (wave >= MT * NT) return;
  const int m0 = (wave / NT) * 16;
  const int n0 = (wave % NT) * 16;
  const int colB = n0 + ln;

  const __bf16* arow = hsbf + (size_t)(m0 + ln) * H_;
  const __bf16* br[E_];
#pragma unroll
  for (int e = 0; e < E_; ++e) br[e] = wobf + ((size_t)e * C_ + colB) * H_;

  v8f acc[E_];
#pragma unroll
  for (int e = 0; e < E_; ++e) acc[e] = zero8();

  for (int k0 = 0; k0 < H_; k0 += 32) {
    ABFrag a;
    a.q[0] = ldg128(arow + k0 + hi * 8);
    a.q[1] = ldg128(arow + k0 + 16 + hi * 8);
    if (k0 + 32 < H_) __builtin_prefetch(arow + k0 + 32, 0, 3);
#pragma unroll
    for (int e = 0; e < E_; ++e) {
      const __bf16* p = br[e] + k0 + hi * 8;
      ABFrag b;
      b.q[0] = ldg128(p);
      b.q[1] = ldg128(p + 16);
      acc[e] = wmma_bf16(a.v, b.v, acc[e]);
    }
  }
#pragma unroll
  for (int v = 0; v < 8; ++v) {
    int m = m0 + v + hi * 8;
    int b = m & (B_ - 1);
    float s = boc[b * C_ + colB];
#pragma unroll
    for (int e = 0; e < E_; ++e) s += coef[b * E_ + e] * acc[e][v];
    out[(size_t)m * C_ + colB] = s;
  }
}

// ------------------------------- launcher ----------------------------------
extern "C" void kernel_launch(void* const* d_in, const int* in_sizes, int n_in,
                              void* d_out, int out_size, void* d_ws, size_t ws_size,
                              hipStream_t stream) {
  const float* x    = (const float*)d_in[0];  // [T,B,I]
  const float* h0   = (const float*)d_in[1];  // [1,B,H]
  const float* c0   = (const float*)d_in[2];  // [1,B,H]
  const float* coef = (const float*)d_in[3];  // [B,E]
  const float* Wi   = (const float*)d_in[4];  // [E,4H,I]
  const float* bi   = (const float*)d_in[5];  // [E,4H]
  const float* Wh   = (const float*)d_in[6];  // [E,4H,H]
  const float* bh   = (const float*)d_in[7];  // [E,4H]
  const float* Wo   = (const float*)d_in[8];  // [E,C,H]
  const float* bo   = (const float*)d_in[9];  // [E,C]
  float* out = (float*)d_out;                 // [T,B,C] ++ hT[B,H] ++ cT[B,H]

  char* ws = (char*)d_ws;
  const size_t OFF_WI  = 0;
  const size_t OFF_WH  = OFF_WI + (size_t)E_ * G_ * I_ * 2;
  const size_t OFF_WO  = OFF_WH + (size_t)E_ * G_ * H_ * 2;
  const size_t OFF_X   = OFF_WO + (size_t)E_ * C_ * H_ * 2;
  const size_t OFF_XI  = OFF_X  + (size_t)T_ * B_ * I_ * 2;
  const size_t OFF_H   = OFF_XI + (size_t)T_ * B_ * G_ * 4;
  const size_t OFF_HS  = OFF_H  + (size_t)B_ * H_ * 2;
  const size_t OFF_C   = OFF_HS + (size_t)T_ * B_ * H_ * 2;
  const size_t OFF_G   = OFF_C  + (size_t)B_ * H_ * 4;
  const size_t OFF_BG  = OFF_G  + (size_t)B_ * G_ * 4;
  const size_t OFF_BO  = OFF_BG + (size_t)B_ * G_ * 4;
  const size_t OFF_BAR = OFF_BO + (size_t)B_ * C_ * 4;

  __bf16* wibf = (__bf16*)(ws + OFF_WI);
  __bf16* whbf = (__bf16*)(ws + OFF_WH);
  __bf16* wobf = (__bf16*)(ws + OFF_WO);
  __bf16* xbf  = (__bf16*)(ws + OFF_X);
  float*  xi   = (float*)(ws + OFF_XI);
  __bf16* hbf  = (__bf16*)(ws + OFF_H);
  __bf16* hsbf = (__bf16*)(ws + OFF_HS);
  float*  cbuf = (float*)(ws + OFF_C);
  float*  gts  = (float*)(ws + OFF_G);
  float*  bg   = (float*)(ws + OFF_BG);
  float*  boc  = (float*)(ws + OFF_BO);
  int*    bar  = (int*)(ws + OFF_BAR);

  // bf16 packing of weights/activations (weights end up L2-resident)
  k_f32_to_bf16<<<4096, 256, 0, stream>>>(Wi, wibf, E_ * G_ * I_);
  k_f32_to_bf16<<<4096, 256, 0, stream>>>(Wh, whbf, E_ * G_ * H_);
  k_f32_to_bf16<<<512,  256, 0, stream>>>(Wo, wobf, E_ * C_ * H_);
  k_f32_to_bf16<<<4096, 256, 0, stream>>>(x,  xbf,  T_ * B_ * I_);

  k_init<<<(B_ * G_ + 255) / 256, 256, 0, stream>>>(coef, bi, bh, bo, h0, c0,
                                                    bg, boc, hbf, cbuf, bar);

  // input projection: 262144 16x16 tiles, 1 tile/wave, 8 waves/block
  k_gemm_xi<<<32768, 256, 0, stream>>>(xbf, wibf, coef, bg, xi);

  // persistent recurrence
  float* outHT = out + (size_t)T_ * B_ * C_;
  float* outCT = outHT + (size_t)B_ * H_;
  k_lstm<<<LSTM_BLOCKS, 256, 0, stream>>>(whbf, coef, xi, hbf, cbuf, gts, hsbf,
                                          outHT, outCT, bar);

  // output projection: 8192 tiles
  k_gemm_out<<<1024, 256, 0, stream>>>(hsbf, wobf, coef, boc, out);
}
